// GraphConvClf3_67327907332512
// MI455X (gfx1250) — compile-verified, matrix operand order checked by reading, not run on hardware
//
#include <hip/hip_runtime.h>
#include <hip/hip_bf16.h>
#include <hip/hip_fp16.h>

// ---------------------------------------------------------------------------
// GraphConvClf on MI455X (gfx1250, wave32, WMMA)
//
// Pipeline per conv layer (dims 3->128->128->256->256):
//   1. convert weights W0/W1 (f32 [dout,din]) -> f16 [dout,Kpad] (zero padded)
//   2. dual WMMA GEMM: h = x@W0^T + b0 (f32), v1h = x@W1^T + b1 (f16)
//      one wave per 16x16 tile, shared A operand, 2x v_wmma_f32_16x16x32_f16
//   3. edge scatter: h[recv] += v1h[send] (global_atomic_add_f32, f16 gather
//      halves the gather traffic of the bandwidth-bound scatter phase)
//   4. BatchNorm batch stats (one-pass sum/sumsq channel reduction) + ReLU, cast to f16
// Then per-mesh mean pool (LDS ds_add_f32 partials -> global atomics) and a
// tiny f32 MLP trunk + 4 heads into d_out (448 floats).
// ---------------------------------------------------------------------------

typedef __attribute__((ext_vector_type(16))) _Float16 v16h;
typedef __attribute__((ext_vector_type(8)))  _Float16 v8h;
typedef __attribute__((ext_vector_type(4)))  _Float16 v4h;
typedef __attribute__((ext_vector_type(8)))  float    v8f;

#define N_MESH 32

// -------------------------------------------------------------------- pack
__global__ void pack_verts(const float* __restrict__ verts,
                           _Float16* __restrict__ x0, int V) {
  int v = blockIdx.x * blockDim.x + threadIdx.x;
  if (v >= V) return;
#pragma unroll
  for (int k = 0; k < 32; ++k)
    x0[(size_t)v * 32 + k] = (_Float16)(k < 3 ? verts[v * 3 + k] : 0.0f);
}

__global__ void convert_w(const float* __restrict__ W,
                          _Float16* __restrict__ Wh,
                          int dout, int din, int Kp) {
  int i = blockIdx.x * blockDim.x + threadIdx.x;
  if (i >= dout * Kp) return;
  int r = i / Kp, c = i % Kp;
  Wh[i] = (_Float16)(c < din ? W[(size_t)r * din + c] : 0.0f);
}

// -------------------------------------------------------------------- GEMM
// x:  f16 [V, K]   (K multiple of 32, rows 64B aligned)
// w0/w1: f16 [N, K] (row n = output channel n; B column n == W row n)
// out0 = x@w0^T + b0  (f32)   out1 = x@w1^T + b1  (f16)
__global__ void __launch_bounds__(256)
gemm_dual_wmma(const _Float16* __restrict__ x,
               const _Float16* __restrict__ w0,
               const _Float16* __restrict__ w1,
               const float* __restrict__ b0,
               const float* __restrict__ b1,
               float* __restrict__ out0,
               _Float16* __restrict__ out1,
               int V, int K, int N) {
  const int lane   = threadIdx.x & 31;
  const int wave   = threadIdx.x >> 5;
  const int nTiles = N >> 4;
  const int mTiles = V >> 4;
  int tile = blockIdx.x * (blockDim.x >> 5) + wave;
  if (tile >= mTiles * nTiles) return;
  const int mTile = tile / nTiles;
  const int nTile = tile % nTiles;

  const int r    = lane & 15;          // row (A) / col (B) within tile
  const int koff = (lane & 16) ? 8 : 0; // ISA K striping per lane half

  const _Float16* xr  = x  + (size_t)(mTile * 16 + r) * K + koff;
  const _Float16* w0r = w0 + (size_t)(nTile * 16 + r) * K + koff;
  const _Float16* w1r = w1 + (size_t)(nTile * 16 + r) * K + koff;

  v8f acc0 = {};
  v8f acc1 = {};
  for (int k = 0; k < K; k += 32) {
    if (k + 64 < K) __builtin_prefetch(xr + k + 64, 0, 1); // global_prefetch_b8
    v8h alo  = *(const v8h*)(xr  + k);
    v8h ahi  = *(const v8h*)(xr  + k + 16);
    v8h b0lo = *(const v8h*)(w0r + k);
    v8h b0hi = *(const v8h*)(w0r + k + 16);
    v8h b1lo = *(const v8h*)(w1r + k);
    v8h b1hi = *(const v8h*)(w1r + k + 16);
    v16h a   = __builtin_shufflevector(alo,  ahi,  0,1,2,3,4,5,6,7,8,9,10,11,12,13,14,15);
    v16h bb0 = __builtin_shufflevector(b0lo, b0hi, 0,1,2,3,4,5,6,7,8,9,10,11,12,13,14,15);
    v16h bb1 = __builtin_shufflevector(b1lo, b1hi, 0,1,2,3,4,5,6,7,8,9,10,11,12,13,14,15);
    acc0 = __builtin_amdgcn_wmma_f32_16x16x32_f16(false, a, false, bb0,
                                                  (short)0, acc0, false, false);
    acc1 = __builtin_amdgcn_wmma_f32_16x16x32_f16(false, a, false, bb1,
                                                  (short)0, acc1, false, false);
  }

  const int   col   = nTile * 16 + r;
  const float bias0 = b0[col];
  const float bias1 = b1[col];
  const int   mbase = mTile * 16 + ((lane & 16) ? 8 : 0);
#pragma unroll
  for (int rr = 0; rr < 8; ++rr) {
    size_t o = (size_t)(mbase + rr) * N + col;
    out0[o] = acc0[rr] + bias0;
    out1[o] = (_Float16)(acc1[rr] + bias1);
  }
}

// ---------------------------------------------------------------- scatter
// h[recv,:] += v1h[send,:] for both directions of each edge.
__global__ void edge_scatter(const int* __restrict__ edges,
                             const _Float16* __restrict__ v1,
                             float* __restrict__ h, int nE, int N) {
  int perEdge = N >> 2; // threads per directed edge, 4 channels each
  int gidx = blockIdx.x * blockDim.x + threadIdx.x;
  int e = gidx / perEdge;
  if (e >= 2 * nE) return;
  int c4 = (gidx % perEdge) << 2;
  int recv, send;
  if (e < nE) { recv = edges[2 * e];          send = edges[2 * e + 1]; }
  else        { int e2 = e - nE;
                recv = edges[2 * e2 + 1];     send = edges[2 * e2]; }
  v4h vv = *(const v4h*)(v1 + (size_t)send * N + c4);
  float* hp = h + (size_t)recv * N + c4;
  atomicAdd(hp + 0, (float)vv[0]);
  atomicAdd(hp + 1, (float)vv[1]);
  atomicAdd(hp + 2, (float)vv[2]);
  atomicAdd(hp + 3, (float)vv[3]);
}

// -------------------------------------------------------------------- BN
__global__ void bn_stats(const float* __restrict__ h,
                         float* __restrict__ sums, float* __restrict__ sqs,
                         int V, int N) {
  int c = blockIdx.x * blockDim.x + threadIdx.x;
  if (c >= N) return;
  float s = 0.f, q = 0.f;
  for (int v = blockIdx.y; v < V; v += gridDim.y) {
    float x = h[(size_t)v * N + c];
    s += x; q += x * x;
  }
  atomicAdd(&sums[c], s);
  atomicAdd(&sqs[c], q);
}

__global__ void bn_finalize(const float* __restrict__ sums,
                            const float* __restrict__ sqs,
                            const float* __restrict__ gamma,
                            const float* __restrict__ beta,
                            float* __restrict__ scale,
                            float* __restrict__ shift, int N, float invV) {
  int c = blockIdx.x * blockDim.x + threadIdx.x;
  if (c >= N) return;
  float mu  = sums[c] * invV;
  float var = sqs[c] * invV - mu * mu;
  float sc  = gamma[c] * rsqrtf(var + 1e-5f);
  scale[c] = sc;
  shift[c] = beta[c] - mu * sc;
}

__global__ void bn_relu_cast(const float* __restrict__ h,
                             const float* __restrict__ scale,
                             const float* __restrict__ shift,
                             _Float16* __restrict__ xo, int V, int N) {
  size_t i = (size_t)blockIdx.x * blockDim.x + threadIdx.x;
  if (i >= (size_t)V * N) return;
  int c = (int)(i % N);
  float y = fmaxf(fmaf(h[i], scale[c], shift[c]), 0.f);
  xo[i] = (_Float16)y;
}

// -------------------------------------------------------------------- pool
__global__ void count_kernel(const int* __restrict__ vidx,
                             float* __restrict__ counts, int V) {
  __shared__ float lc[N_MESH];
  if (threadIdx.x < N_MESH) lc[threadIdx.x] = 0.f;
  __syncthreads();
  int v = blockIdx.x * blockDim.x + threadIdx.x;
  if (v < V) atomicAdd(&lc[vidx[v]], 1.f);          // ds_add_f32
  __syncthreads();
  if (threadIdx.x < N_MESH && lc[threadIdx.x] != 0.f)
    atomicAdd(&counts[threadIdx.x], lc[threadIdx.x]);
}

__global__ void __launch_bounds__(256)
pool_kernel(const _Float16* __restrict__ x, const int* __restrict__ vidx,
            float* __restrict__ pooled, int V, int N) {
  __shared__ float acc[N_MESH * 64];
  int tid = threadIdx.x;
  int v   = blockIdx.x * 256 + tid;
  int mesh = (v < V) ? vidx[v] : -1;
  for (int cb = 0; cb < N; cb += 64) {
    for (int i = tid; i < N_MESH * 64; i += 256) acc[i] = 0.f;
    __syncthreads();
    if (mesh >= 0) {
      for (int c = 0; c < 64; ++c)
        atomicAdd(&acc[mesh * 64 + c], (float)x[(size_t)v * N + cb + c]);
    }
    __syncthreads();
    for (int i = tid; i < N_MESH * 64; i += 256) {
      float a = acc[i];
      if (a != 0.f) atomicAdd(&pooled[(size_t)(i / 64) * N + cb + (i % 64)], a);
    }
    __syncthreads();
  }
}

__global__ void pool_div(float* __restrict__ pooled,
                         const float* __restrict__ counts, int N) {
  int i = blockIdx.x * blockDim.x + threadIdx.x;
  if (i >= N_MESH * N) return;
  pooled[i] /= fmaxf(counts[i / N], 1.f);
}

// -------------------------------------------------------------------- MLP
__global__ void linear_kernel(const float* __restrict__ x,
                              const float* __restrict__ W,
                              const float* __restrict__ b,
                              float* __restrict__ y,
                              int B, int Din, int Dout, int doRelu) {
  int t = blockIdx.x * blockDim.x + threadIdx.x;
  if (t >= B * Dout) return;
  int row = t / Dout, o = t % Dout;
  const float* xr = x + (size_t)row * Din;
  const float* wr = W + (size_t)o * Din;
  float acc = b[o];
  for (int k = 0; k < Din; ++k) acc = fmaf(xr[k], wr[k], acc);
  y[t] = doRelu ? fmaxf(acc, 0.f) : acc;
}

// ==================================================================== host
extern "C" void kernel_launch(void* const* d_in, const int* in_sizes, int n_in,
                              void* d_out, int out_size, void* d_ws, size_t ws_size,
                              hipStream_t stream) {
  const float* verts = (const float*)d_in[0];
  const int*   edges = (const int*)d_in[1];
  const int*   vidx  = (const int*)d_in[2];
  const int V  = in_sizes[0] / 3;
  const int nE = in_sizes[1] / 2;

  const int dims[5] = {3, 128, 128, 256, 256};

  // --- workspace carve-out (aligned to 256B) ---
  char* ws = (char*)d_ws;
  size_t off = 0;
  auto alloc = [&](size_t bytes) -> char* {
    char* p = ws + off;
    off = (off + bytes + 255) & ~(size_t)255;
    return p;
  };
  _Float16* xh   = (_Float16*)alloc((size_t)V * 256 * sizeof(_Float16)); // activations (f16)
  float*    h    = (float*)   alloc((size_t)V * 256 * sizeof(float));    // pre-BN accum (f32)
  _Float16* v1h  = (_Float16*)alloc((size_t)V * 256 * sizeof(_Float16)); // neighbor feats (f16)
  _Float16* wh0  = (_Float16*)alloc(256 * 256 * sizeof(_Float16));
  _Float16* wh1  = (_Float16*)alloc(256 * 256 * sizeof(_Float16));
  float* sums    = (float*)alloc(256 * sizeof(float));
  float* sqs     = (float*)alloc(256 * sizeof(float));
  float* scale   = (float*)alloc(256 * sizeof(float));
  float* shift   = (float*)alloc(256 * sizeof(float));
  float* pooled  = (float*)alloc(N_MESH * 256 * sizeof(float));
  float* counts  = (float*)alloc(N_MESH * sizeof(float));
  float* trunk   = (float*)alloc(N_MESH * 512 * sizeof(float));
  float* hhead   = (float*)alloc(N_MESH * 256 * sizeof(float));

  // --- layer-1 input: pad K 3 -> 32, cast f16 ---
  pack_verts<<<(V + 255) / 256, 256, 0, stream>>>(verts, xh, V);

  for (int L = 0; L < 4; ++L) {
    const int din  = dims[L];
    const int dout = dims[L + 1];
    const int Kp   = (din < 32) ? 32 : din;
    const float* W0    = (const float*)d_in[3 + 6 * L + 0];
    const float* b0    = (const float*)d_in[3 + 6 * L + 1];
    const float* W1    = (const float*)d_in[3 + 6 * L + 2];
    const float* b1    = (const float*)d_in[3 + 6 * L + 3];
    const float* gamma = (const float*)d_in[3 + 6 * L + 4];
    const float* beta  = (const float*)d_in[3 + 6 * L + 5];

    const int wcount = dout * Kp;
    convert_w<<<(wcount + 255) / 256, 256, 0, stream>>>(W0, wh0, dout, din, Kp);
    convert_w<<<(wcount + 255) / 256, 256, 0, stream>>>(W1, wh1, dout, din, Kp);

    const int tiles = (V / 16) * (dout / 16);
    gemm_dual_wmma<<<(tiles + 7) / 8, 256, 0, stream>>>(
        xh, wh0, wh1, b0, b1, h, v1h, V, Kp, dout);

    const long long sthreads = 2LL * nE * (dout / 4);
    edge_scatter<<<(int)((sthreads + 255) / 256), 256, 0, stream>>>(
        edges, v1h, h, nE, dout);

    hipMemsetAsync(sums, 0, 256 * sizeof(float), stream);
    hipMemsetAsync(sqs,  0, 256 * sizeof(float), stream);
    dim3 sg((dout + 127) / 128, 512);
    bn_stats<<<sg, 128, 0, stream>>>(h, sums, sqs, V, dout);
    bn_finalize<<<(dout + 127) / 128, 128, 0, stream>>>(
        sums, sqs, gamma, beta, scale, shift, dout, 1.0f / (float)V);

    const size_t tot = (size_t)V * dout;
    bn_relu_cast<<<(int)((tot + 255) / 256), 256, 0, stream>>>(
        h, scale, shift, xh, V, dout);
  }

  // --- per-mesh mean pool ---
  hipMemsetAsync(pooled, 0, N_MESH * 256 * sizeof(float), stream);
  hipMemsetAsync(counts, 0, N_MESH * sizeof(float), stream);
  count_kernel<<<(V + 255) / 256, 256, 0, stream>>>(vidx, counts, V);
  pool_kernel<<<(V + 255) / 256, 256, 0, stream>>>(xh, vidx, pooled, V, 256);
  pool_div<<<(N_MESH * 256 + 255) / 256, 256, 0, stream>>>(pooled, counts, 256);

  // --- MLP trunk + heads ---
  const float* Wt = (const float*)d_in[27];
  const float* bt = (const float*)d_in[28];
  linear_kernel<<<(N_MESH * 512 + 255) / 256, 256, 0, stream>>>(
      pooled, Wt, bt, trunk, N_MESH, 256, 512, 1);

  float* out = (float*)d_out;
  const int headOut[4] = {3, 2, 4, 5};
  int ooff = 0;
  for (int i = 0; i < 4; ++i) {
    const float* W1h = (const float*)d_in[29 + 4 * i];
    const float* b1h = (const float*)d_in[30 + 4 * i];
    const float* W2h = (const float*)d_in[31 + 4 * i];
    const float* b2h = (const float*)d_in[32 + 4 * i];
    linear_kernel<<<(N_MESH * 256 + 255) / 256, 256, 0, stream>>>(
        trunk, W1h, b1h, hhead, N_MESH, 512, 256, 1);
    const int dn = headOut[i];
    linear_kernel<<<(N_MESH * dn + 255) / 256, 256, 0, stream>>>(
        hhead, W2h, b2h, out + ooff, N_MESH, 256, dn, 0);
    ooff += N_MESH * dn;
  }
}
